// Decoder_55448027791455
// MI455X (gfx1250) — matrix operand order repaired and checked by reading, not compiled, not run on hardware
//
#include <hip/hip_runtime.h>
#include <hip/hip_bf16.h>
#include <math.h>

// Problem dims (fixed by the reference)
#define V_  32000
#define E_  1024
#define H_  1024
#define A_  1024
#define B_  64
#define S_  512
#define XC_ (E_ + 2 * H_)   // [emb | ctx | h0] = 3072

typedef __attribute__((ext_vector_type(16))) __bf16 v16bf;
typedef __attribute__((ext_vector_type(8)))  __bf16 v8bf;
typedef __attribute__((ext_vector_type(4)))  __bf16 v4bf;
typedef __attribute__((ext_vector_type(8)))  float  v8f;
typedef __attribute__((ext_vector_type(2)))  float  v2f;
typedef __attribute__((ext_vector_type(4)))  unsigned v4u;
typedef __attribute__((ext_vector_type(4)))  int    v4i;
typedef __attribute__((ext_vector_type(8)))  int    v8i;

// ---- CDNA5 feature detection -----------------------------------------------
#if defined(__has_builtin)
# if __has_builtin(__builtin_amdgcn_tensor_load_to_lds)
#  define HAVE_TDM 1
# endif
# if __has_builtin(__builtin_amdgcn_tanhf)
#  define TANHF(x) __builtin_amdgcn_tanhf(x)      // native v_tanh_f32
# endif
# if __has_builtin(__builtin_amdgcn_s_wait_tensorcnt)
#  define WAIT_TENSOR() __builtin_amdgcn_s_wait_tensorcnt(0)
# else
#  define WAIT_TENSOR() asm volatile("s_wait_tensorcnt 0x0" ::: "memory")
# endif
#endif
#ifndef HAVE_TDM
# define HAVE_TDM 0
#endif
#ifndef TANHF
# define TANHF(x) (1.0f - 2.0f / (__expf(2.0f * (x)) + 1.0f))  // branch-free fallback
#endif

static __device__ __forceinline__ __bf16 f2bf(float f) {
  unsigned u = __builtin_bit_cast(unsigned, f);
  u += 0x7FFFu + ((u >> 16) & 1u);             // round-to-nearest-even
  unsigned short s = (unsigned short)(u >> 16);
  return __builtin_bit_cast(__bf16, s);
}

static __device__ __forceinline__ float sigf(float x) {
  return 0.5f * TANHF(0.5f * x) + 0.5f;        // branch-free sigmoid via v_tanh_f32
}

// ---------------------------------------------------------------------------
// K1: t1[b][a] = h0[b]·Wa[a] + b_wa[a] + b_ua[a]     (64 x 1024, tiny)
// ---------------------------------------------------------------------------
__global__ void k_prep_t1(const float* __restrict__ h0, const float* __restrict__ Wa,
                          const float* __restrict__ bwa, const float* __restrict__ bua,
                          float* __restrict__ t1) {
  __shared__ float hs[H_];
  const int tid = threadIdx.x;                 // 256
  const int a   = blockIdx.x * 256 + tid;      // gridDim.x = 4
  const int b   = blockIdx.y;                  // 64
  for (int i = tid; i < H_; i += 256) hs[i] = h0[b * H_ + i];
  __syncthreads();
  const float4* wr = (const float4*)(Wa + (size_t)a * H_);
  float acc = 0.f;
  #pragma unroll 4
  for (int h4 = 0; h4 < H_ / 4; ++h4) {
    float4 w = wr[h4];
    acc += w.x * hs[4 * h4 + 0] + w.y * hs[4 * h4 + 1]
         + w.z * hs[4 * h4 + 2] + w.w * hs[4 * h4 + 3];
  }
  t1[(size_t)b * A_ + a] = acc + bwa[a] + bua[a];
}

// ---------------------------------------------------------------------------
// K2: Ua f32 -> *transposed* bf16 scratch uabT[k][a]  (2 MB, written once).
// The transpose here makes every per-a0 B tile a contiguous-row 2D tensor,
// which the Tensor Data Mover can DMA straight into LDS in the WMMA B layout.
// ---------------------------------------------------------------------------
__global__ void k_cvt_ua(const float* __restrict__ Ua, __bf16* __restrict__ uabT) {
  int i = blockIdx.x * 256 + threadIdx.x;      // over A_*H_ elements
  int a = i >> 10;                             // row of Ua   (H_ = 1024)
  int k = i & (H_ - 1);                        // col of Ua
  uabT[(size_t)k * A_ + a] = f2bf(Ua[i]);      // coalesced read, strided write (one-time)
}

// ---------------------------------------------------------------------------
// K3: fused attention-score GEMM (the 69-GFLOP hot spot).
//   scores[r] = b_va + sum_a va[a] * tanh(t1[b(r)][a] + enc[r]·Ua[a])
// Block = 128 threads (4 waves), 64 enc rows staged once in LDS as bf16.
// Per a-chunk of 16: TDM loads the Ua^T tile (tile_dim0=16 elems x 1024 rows,
// stride A_) into LDS, then K-loop of v_wmma_f32_16x16x32_bf16; epilogue folds
// native-tanh + va-dot into per-lane partials. LDS: 128KB enc + 32KB tile.
// ---------------------------------------------------------------------------
__global__ void k_scores(const float* __restrict__ enc, const float* __restrict__ t1,
                         const __bf16* __restrict__ uabT, const float* __restrict__ va,
                         const float* __restrict__ bva, float* __restrict__ scores) {
  __shared__ __align__(16) __bf16 encb[64 * H_];   // 64 rows, bf16 (128 KB)
  __shared__ __align__(16) __bf16 uat[H_ * 16];    // B tile [k][n]   (32 KB)
  const int tid  = threadIdx.x;                    // 0..127
  const int lane = tid & 31;
  const int wave = tid >> 5;                       // 0..3
  const int r0   = blockIdx.x * 64;                // 512 blocks * 64 rows

  // Stage 64 enc rows (f32 -> bf16), coalesced float4 reads, b64 LDS writes.
  for (int i = tid; i < 64 * (H_ / 4); i += 128) {
    float4 v = ((const float4*)enc)[(size_t)r0 * (H_ / 4) + i];
    v4bf o; o[0] = f2bf(v.x); o[1] = f2bf(v.y); o[2] = f2bf(v.z); o[3] = f2bf(v.w);
    *(v4bf*)&encb[i * 4] = o;
  }

#if HAVE_TDM
  const unsigned lds_uat = (unsigned)(size_t)(void*)&uat[0];   // LDS byte offset
#endif

  const int mrow = lane & 15;
  const int half = lane >> 4;
  const int wrow = wave * 16;
  float ps[8] = {0, 0, 0, 0, 0, 0, 0, 0};

  for (int a0 = 0; a0 < A_; a0 += 16) {
    __syncthreads();                 // previous tile fully consumed
#if HAVE_TDM
    if (wave == 0) {
      // D# group 0: count=1(valid), lds_addr, 57b global addr, type=2
      unsigned long long ga = (unsigned long long)(size_t)(uabT + a0);
      v4u g0 = { 1u, lds_uat, (unsigned)ga,
                 ((unsigned)(ga >> 32) & 0x01FFFFFFu) | 0x80000000u };
      // D# group 1: data_size=1(2B); tensor_dim0=16, tensor_dim1=1024;
      // tile_dim0=16, tile_dim1=1024; tensor_dim0_stride=A_(=1024 elems)
      v8i g1 = { (int)0x00010000,     // wg_mask=0 | data_size=1<<16
                 (int)(16u << 16),    // tensor_dim0[15:0]
                 (int)(1024u << 16),  // tensor_dim0[31:16]=0 | tensor_dim1[15:0]
                 (int)(16u << 16),    // tensor_dim1[31:16]=0 | tile_dim0
                 1024,                // tile_dim1 | tile_dim2=0
                 1024,                // tensor_dim0_stride[31:0]
                 0, 0 };              // stride hi / tensor_dim1_stride (2D)
      v4i gz = {0, 0, 0, 0};
#if defined(__clang_major__) && (__clang_major__ >= 23)
      v8i gz8 = {0, 0, 0, 0, 0, 0, 0, 0};
      __builtin_amdgcn_tensor_load_to_lds(g0, g1, gz, gz, gz8, 0);
#else
      __builtin_amdgcn_tensor_load_to_lds(g0, g1, gz, gz, 0);
#endif
      WAIT_TENSOR();                 // s_wait_tensorcnt 0
    }
#else
    // Fallback: uabT rows are contiguous 32B chunks -> simple vector copy.
    for (int i = tid; i < H_ * 2; i += 128) {
      int k = i >> 1, c = i & 1;
      *(v8bf*)&uat[k * 16 + c * 8] =
          *(const v8bf*)&uabT[(size_t)k * A_ + a0 + c * 8];
    }
#endif
    __syncthreads();                 // tile visible to all 4 waves

    v8f acc = {};
    #pragma unroll 4
    for (int k0 = 0; k0 < H_; k0 += 32) {
      // A fragment: lane holds row (wrow+mrow), K = k0 + 16*(t/8) + 8*half + t%8
      const __bf16* ap = &encb[(wrow + mrow) * H_ + k0 + half * 8];
      v8bf alo = *(const v8bf*)ap;            // ds_load_b128
      v8bf ahi = *(const v8bf*)(ap + 16);     // ds_load_b128
      // B fragment: lane holds col N = lane%16, K = k0 + 16*(t/8) + 8*half + t%8
      const __bf16* bp = &uat[(k0 + half * 8) * 16 + mrow];
      v8bf blo, bhi;
      #pragma unroll
      for (int i = 0; i < 8; ++i) { blo[i] = bp[i * 16]; bhi[i] = bp[(16 + i) * 16]; }
      v16bf af, bf;
      #pragma unroll
      for (int i = 0; i < 8; ++i) {
        af[i] = alo[i]; af[8 + i] = ahi[i];
        bf[i] = blo[i]; bf[8 + i] = bhi[i];
      }
      acc = __builtin_amdgcn_wmma_f32_16x16x32_bf16(
          false, af, false, bf, (short)0, acc, false, false);
    }

    // Epilogue: D[m][n] at reg r: M = r + 8*half, N = lane%16 (branch-free tanh)
    const int n = lane & 15;
    const float van = va[a0 + n];
    #pragma unroll
    for (int r = 0; r < 8; ++r) {
      int row = r0 + wrow + r + 8 * half;     // global (s*B+b) row
      int b   = row & (B_ - 1);
      float t1v = t1[(size_t)b * A_ + a0 + n];
      ps[r] += van * TANHF(acc[r] + t1v);
    }
  }

  // Reduce partials across each 16-lane half (wave32 shuffles).
  const float bv = bva[0];
  #pragma unroll
  for (int r = 0; r < 8; ++r) {
    float v = ps[r];
    v += __shfl_xor(v, 8, 32);
    v += __shfl_xor(v, 4, 32);
    v += __shfl_xor(v, 2, 32);
    v += __shfl_xor(v, 1, 32);
    if ((lane & 15) == 0)
      scores[r0 + wrow + r + 8 * half] = v + bv;
  }
}

// ---------------------------------------------------------------------------
// K4: per-batch softmax over S + context sum + assemble xcat = [emb|ctx|h0].
// enc is re-read here: 134 MB fp32 sits in the 192 MB L2 after k_scores.
// ---------------------------------------------------------------------------
__global__ void k_attn_ctx(const float* __restrict__ scores, const float* __restrict__ enc,
                           const float* __restrict__ emb, const int* __restrict__ x,
                           const float* __restrict__ h0, float* __restrict__ xcat) {
  const int b = blockIdx.x;                    // 64 blocks, 256 threads
  const int tid = threadIdx.x;
  __shared__ float w[S_];
  __shared__ float red[256];

  float m = -INFINITY;
  for (int s = tid; s < S_; s += 256) {
    float v = scores[s * B_ + b];
    w[s] = v; m = fmaxf(m, v);
  }
  red[tid] = m; __syncthreads();
  for (int o = 128; o > 0; o >>= 1) { if (tid < o) red[tid] = fmaxf(red[tid], red[tid + o]); __syncthreads(); }
  m = red[0]; __syncthreads();

  float sum = 0.f;
  for (int s = tid; s < S_; s += 256) { float e = __expf(w[s] - m); w[s] = e; sum += e; }
  red[tid] = sum; __syncthreads();
  for (int o = 128; o > 0; o >>= 1) { if (tid < o) red[tid] += red[tid + o]; __syncthreads(); }
  const float inv = 1.0f / red[0];
  __syncthreads();
  for (int s = tid; s < S_; s += 256) w[s] *= inv;
  __syncthreads();

  // ctx[h..h+3] = sum_s w[s] * enc[s][b][h..h+3]
  float4 acc = {0.f, 0.f, 0.f, 0.f};
  for (int s = 0; s < S_; ++s) {
    float ws = w[s];
    float4 e = ((const float4*)enc)[((size_t)s * B_ + b) * (H_ / 4) + tid];
    acc.x += ws * e.x; acc.y += ws * e.y; acc.z += ws * e.z; acc.w += ws * e.w;
  }
  float* xr = xcat + (size_t)b * XC_;
  ((float4*)(xr + E_))[tid] = acc;                                          // ctx
  ((float4*)xr)[tid] = ((const float4*)emb)[(size_t)x[b] * (E_ / 4) + tid]; // emb gather
  ((float4*)(xr + E_ + H_))[tid] = ((const float4*)h0)[(size_t)b * (H_ / 4) + tid]; // h0
}

// ---------------------------------------------------------------------------
// K5: gate GEMM, fp32 WMMA 16x16x4. gates = xcat[64,3072] x [W_ih|W_hh]^T + b.
// Each wave owns one 16-wide N tile, reuses its B-fragment over 4 M tiles,
// so the 50 MB of weights stream from HBM exactly once (memory-bound).
// ---------------------------------------------------------------------------
__global__ void k_gates(const float* __restrict__ xcat, const float* __restrict__ Wih,
                        const float* __restrict__ Whh, const float* __restrict__ bih,
                        const float* __restrict__ bhh, float* __restrict__ gates) {
  const int tid  = threadIdx.x;                // 256 = 8 waves
  const int lane = tid & 31;
  const int wave = tid >> 5;
  const int n0   = (blockIdx.x * 8 + wave) * 16;   // 32 blocks -> 4096 gates
  const int nn   = lane & 15;
  const int half = lane >> 4;

  v8f acc[4] = {{}, {}, {}, {}};
  for (int k0 = 0; k0 < XC_; k0 += 4) {
    // B fragment: element e -> K = k0 + 2e + half, N = nn  (W[n][k])
    v2f bfr;
    {
      int ka = k0 + half, kb = k0 + 2 + half;    // never straddles 2048 (k0 % 4 == 0)
      if (ka < 2048) {
        const float* wr = Wih + (size_t)(n0 + nn) * 2048;
        bfr[0] = wr[ka]; bfr[1] = wr[kb];
      } else {
        const float* wr = Whh + (size_t)(n0 + nn) * 1024 - 2048;
        bfr[0] = wr[ka]; bfr[1] = wr[kb];
      }
    }
    #pragma unroll
    for (int mt = 0; mt < 4; ++mt) {
      // A fragment: element e -> K = k0 + 2*half + e, M = mt*16 + lane%16
      const float* ar = xcat + (size_t)(mt * 16 + nn) * XC_ + k0 + 2 * half;
      v2f afr; afr[0] = ar[0]; afr[1] = ar[1];
      acc[mt] = __builtin_amdgcn_wmma_f32_16x16x4_f32(
          false, afr, false, bfr, (short)0, acc[mt], false, false);
    }
  }
  const float bias = bih[n0 + nn] + bhh[n0 + nn];
  #pragma unroll
  for (int mt = 0; mt < 4; ++mt)
    #pragma unroll
    for (int r = 0; r < 8; ++r) {
      int m = mt * 16 + r + 8 * half;            // batch row
      gates[(size_t)m * (4 * H_) + n0 + nn] = acc[mt][r] + bias;
    }
}

// ---------------------------------------------------------------------------
// K6: LSTM elementwise (PyTorch gate order i,f,g,o) — native v_tanh_f32 paths
// ---------------------------------------------------------------------------
__global__ void k_lstm(const float* __restrict__ gates, const float* __restrict__ c0,
                       float* __restrict__ h1) {
  int idx = blockIdx.x * 256 + threadIdx.x;      // 64*1024
  int b = idx >> 10, h = idx & (H_ - 1);
  const float* g = gates + (size_t)b * (4 * H_);
  float i = sigf(g[h]);
  float f = sigf(g[H_ + h]);
  float gg = TANHF(g[2 * H_ + h]);
  float o = sigf(g[3 * H_ + h]);
  float c1 = f * c0[idx] + i * gg;
  h1[idx] = o * TANHF(c1);
}

// ---------------------------------------------------------------------------
// K7: classifier GEMM logits = h1[64,1024] x Wclf^T[1024,32000] + b_clf.
// Same fp32-WMMA tiling; 131 MB weight read once from HBM (~5.6 us bound).
// ---------------------------------------------------------------------------
__global__ void k_clf(const float* __restrict__ h1, const float* __restrict__ Wclf,
                      const float* __restrict__ bclf, float* __restrict__ out) {
  const int tid  = threadIdx.x;                // 256 = 8 waves
  const int lane = tid & 31;
  const int wave = tid >> 5;
  const int n0   = (blockIdx.x * 8 + wave) * 16;   // 250 blocks -> 32000 vocab
  const int nn   = lane & 15;
  const int half = lane >> 4;
  const float* wr = Wclf + (size_t)(n0 + nn) * H_;

  v8f acc[4] = {{}, {}, {}, {}};
  for (int k0 = 0; k0 < H_; k0 += 4) {
    __builtin_prefetch(wr + k0 + 128, 0, 1);   // global_prefetch_b8 a few lines ahead
    v2f bfr; bfr[0] = wr[k0 + half]; bfr[1] = wr[k0 + 2 + half];
    #pragma unroll
    for (int mt = 0; mt < 4; ++mt) {
      const float* ar = h1 + (size_t)(mt * 16 + nn) * H_ + k0 + 2 * half;
      v2f afr; afr[0] = ar[0]; afr[1] = ar[1];
      acc[mt] = __builtin_amdgcn_wmma_f32_16x16x4_f32(
          false, afr, false, bfr, (short)0, acc[mt], false, false);
    }
  }
  const float bias = bclf[n0 + nn];
  #pragma unroll
  for (int mt = 0; mt < 4; ++mt)
    #pragma unroll
    for (int r = 0; r < 8; ++r) {
      int m = mt * 16 + r + 8 * half;
      out[(size_t)m * V_ + n0 + nn] = acc[mt][r] + bias;
    }
}

// ---------------------------------------------------------------------------
// K8: in-place log_softmax over V per batch row
// ---------------------------------------------------------------------------
__global__ void k_logsoftmax(float* __restrict__ out) {
  const int b = blockIdx.x;                    // 64 blocks, 256 threads
  const int tid = threadIdx.x;
  float* row = out + (size_t)b * V_;
  __shared__ float red[256];

  float m = -INFINITY;
  for (int v = tid; v < V_; v += 256) m = fmaxf(m, row[v]);
  red[tid] = m; __syncthreads();
  for (int o = 128; o > 0; o >>= 1) { if (tid < o) red[tid] = fmaxf(red[tid], red[tid + o]); __syncthreads(); }
  m = red[0]; __syncthreads();

  float s = 0.f;
  for (int v = tid; v < V_; v += 256) s += __expf(row[v] - m);
  red[tid] = s; __syncthreads();
  for (int o = 128; o > 0; o >>= 1) { if (tid < o) red[tid] += red[tid + o]; __syncthreads(); }
  const float lse = m + __logf(red[0]);
  __syncthreads();

  for (int v = tid; v < V_; v += 256) row[v] = row[v] - lse;
}

// ---------------------------------------------------------------------------
extern "C" void kernel_launch(void* const* d_in, const int* in_sizes, int n_in,
                              void* d_out, int out_size, void* d_ws, size_t ws_size,
                              hipStream_t stream) {
  const int*   x    = (const int*)  d_in[0];
  const float* enc  = (const float*)d_in[1];
  const float* h0   = (const float*)d_in[2];
  const float* c0   = (const float*)d_in[3];
  const float* Wa   = (const float*)d_in[4];
  const float* bwa  = (const float*)d_in[5];
  const float* Ua   = (const float*)d_in[6];
  const float* bua  = (const float*)d_in[7];
  const float* va   = (const float*)d_in[8];
  const float* bva  = (const float*)d_in[9];
  const float* emb  = (const float*)d_in[10];
  const float* Wih  = (const float*)d_in[11];
  const float* Whh  = (const float*)d_in[12];
  const float* bih  = (const float*)d_in[13];
  const float* bhh  = (const float*)d_in[14];
  const float* Wclf = (const float*)d_in[15];
  const float* bclf = (const float*)d_in[16];
  float* out = (float*)d_out;

  // Workspace carve-up (~4.4 MB)
  char* p = (char*)d_ws;
  float*  t1     = (float*)p;  p += (size_t)B_ * A_ * 4;        // 256 KB
  __bf16* uabT   = (__bf16*)p; p += (size_t)A_ * H_ * 2;        // 2 MB (transposed)
  float*  scores = (float*)p;  p += (size_t)S_ * B_ * 4;        // 128 KB
  float*  xcat   = (float*)p;  p += (size_t)B_ * XC_ * 4;       // 768 KB
  float*  gates  = (float*)p;  p += (size_t)B_ * 4 * H_ * 4;    // 1 MB
  float*  h1     = (float*)p;  p += (size_t)B_ * H_ * 4;        // 256 KB

  k_prep_t1   <<<dim3(A_ / 256, B_), 256, 0, stream>>>(h0, Wa, bwa, bua, t1);
  k_cvt_ua    <<<(A_ * H_) / 256, 256, 0, stream>>>(Ua, uabT);
  k_scores    <<<(S_ * B_) / 64, 128, 0, stream>>>(enc, t1, uabT, va, bva, scores);
  k_attn_ctx  <<<B_, 256, 0, stream>>>(scores, enc, emb, x, h0, xcat);
  k_gates     <<<(4 * H_ / 16) / 8, 256, 0, stream>>>(xcat, Wih, Whh, bih, bhh, gates);
  k_lstm      <<<(B_ * H_) / 256, 256, 0, stream>>>(gates, c0, h1);
  k_clf       <<<(V_ / 16) / 8, 256, 0, stream>>>(h1, Wclf, bclf, out);
  k_logsoftmax<<<B_, 256, 0, stream>>>(out);
}
